// RelationalBBoxNetworkEncoder_48962627174544
// MI455X (gfx1250) — matrix operand order repaired and checked by reading, not compiled, not run on hardware
//
#include <hip/hip_runtime.h>
#include <hip/hip_bf16.h>

typedef __attribute__((ext_vector_type(16))) _Float16 v16h;
typedef __attribute__((ext_vector_type(8)))  float    v8f;
typedef __attribute__((ext_vector_type(8)))  _Float16 v8h;

#define CD   1280      // roi feature dim C
#define HD   1280      // hidden dim H
#define BM   1024      // B*M rows
#define MT   32        // row-tile (boxes per image)
#define RCH  160       // 16B chunks per 1280-half row

#define WMMA_F16(A, B, C) \
  __builtin_amdgcn_wmma_f32_16x16x32_f16(false, (A), false, (B), (short)0, (C), false, false)

union FragU { v16h v; float4 f4[2]; };

// B fragment (32x16 f16) from transposed weights [n][k]:
// lane l(<16) = col l, K 0..15 contiguous; lane l+16 = col l, K 16..31.
// p = colbase + kb + (hi ? 16 : 0)
__device__ __forceinline__ v16h load_b_frag(const _Float16* p) {
  FragU u;
  u.f4[0] = *reinterpret_cast<const float4*>(p);
  u.f4[1] = *reinterpret_cast<const float4*>(p + 8);
  return u.v;
}

// A fragment (16x32 f16) from an XOR-swizzled LDS tile (row stride 1280 halves;
// 16B chunk c of row r is stored at chunk c^(r&15)).  Here each lane
// takes K {0..15} (lanes<16) / {16..31} (lanes>=16), as B does: chunks c0, c0+1
// with c0 = kb/8 + 2*hi.
__device__ __forceinline__ v16h load_a_swz(const _Float16* rowbase, int swz, int c0) {
  FragU u;
  u.f4[0] = *reinterpret_cast<const float4*>(rowbase + ((c0 ^ swz) << 3));
  u.f4[1] = *reinterpret_cast<const float4*>(rowbase + (((c0 + 1) ^ swz) << 3));
  return u.v;
}

// ---------------- prep kernels ----------------
__global__ void k_cvt_f16(const float* __restrict__ in, _Float16* __restrict__ out, int n4) {
  int i = blockIdx.x * blockDim.x + threadIdx.x;
  if (i >= n4) return;
  float4 f = reinterpret_cast<const float4*>(in)[i];
  v8h dummy; (void)dummy;
  __attribute__((ext_vector_type(4))) _Float16 h;
  h[0] = (_Float16)f.x; h[1] = (_Float16)f.y; h[2] = (_Float16)f.z; h[3] = (_Float16)f.w;
  *reinterpret_cast<decltype(h)*>(out + i * 4) = h;
}

// WT[n*K + k] = (f16) W[k*N + n]   (coalesced writes)
__global__ void k_transpose_f16(const float* __restrict__ W, _Float16* __restrict__ WT,
                                int K, int N) {
  int idx = blockIdx.x * blockDim.x + threadIdx.x;
  if (idx >= K * N) return;
  int n = idx / K, k = idx % K;
  WT[idx] = (_Float16)W[(size_t)k * N + n];
}

__global__ void k_zero(float* __restrict__ p, int n) {
  int i = blockIdx.x * blockDim.x + threadIdx.x;
  if (i < n) p[i] = 0.0f;
}

// ---------------- stage 1: a = roi @ W1 halves ----------------
// grid = 64: blockIdx = (half<<5) | rowblock. 8 waves; wave w owns cols [w*160, w*160+160)
__global__ __launch_bounds__(256)
void k_gemm1(const _Float16* __restrict__ roih,  // [BM][CD] f16
             const _Float16* __restrict__ W1T,   // [HD][2*CD] f16 (n-major)
             float* __restrict__ aI, float* __restrict__ aJ) {
  extern __shared__ _Float16 sA[];               // [MT][CD] swizzled, 80KB
  const int rb = blockIdx.x & 31;
  const int half = blockIdx.x >> 5;
  const int tid = threadIdx.x;

  { // stage 32 rows into LDS with chunk swizzle
    const uint4* src = reinterpret_cast<const uint4*>(roih + (size_t)rb * MT * CD);
    uint4* dst = reinterpret_cast<uint4*>(sA);
    for (int i = tid; i < MT * RCH; i += 256) {
      int q = i / RCH, c = i % RCH;
      dst[q * RCH + (c ^ (q & 15))] = src[i];
    }
  }
  __syncthreads();

  const int wave = tid >> 5, lane = tid & 31;
  const int l16 = lane & 15, hi = lane >> 4;
  float* aout = half ? aJ : aI;
  const _Float16* a0base = sA + l16 * CD;
  const _Float16* a1base = a0base + 16 * CD;

  for (int g = 0; g < 5; ++g) {                  // 2m x 2n register blocking
    const int nb0 = wave * 160 + g * 32, nb1 = nb0 + 16;
    v8f acc00 = {}, acc01 = {}, acc10 = {}, acc11 = {};
    const _Float16* b0p = W1T + (size_t)(nb0 + l16) * (2 * CD) + half * CD + (hi << 4);
    const _Float16* b1p = b0p + (size_t)16 * (2 * CD);
#pragma unroll 2
    for (int kb = 0; kb < CD; kb += 32) {
      const int c0 = (kb >> 3) + (hi << 1);
      v16h a0 = load_a_swz(a0base, l16, c0);
      v16h a1 = load_a_swz(a1base, l16, c0);
      v16h b0 = load_b_frag(b0p + kb);
      v16h b1 = load_b_frag(b1p + kb);
      acc00 = WMMA_F16(a0, b0, acc00);
      acc01 = WMMA_F16(a0, b1, acc01);
      acc10 = WMMA_F16(a1, b0, acc10);
      acc11 = WMMA_F16(a1, b1, acc11);
    }
    // D layout: VGPR r, lanes<16 -> M=r ; lanes>=16 -> M=r+8 ; N = lane&15
    float* o0 = aout + (size_t)(rb * MT + hi * 8) * HD + l16;
    float* o1 = aout + (size_t)(rb * MT + 16 + hi * 8) * HD + l16;
#pragma unroll
    for (int r = 0; r < 8; ++r) {
      o0[r * HD + nb0] = acc00[r];  o0[r * HD + nb1] = acc01[r];
      o1[r * HD + nb0] = acc10[r];  o1[r * HD + nb1] = acc11[r];
    }
  }
}

// ---------------- stage 2: fused pairwise + 2 GEMMs + reduction ----------------
// grid = 1024 : blockIdx = (b<<5) | p.  h1 tile = relu(a_i[b,q,:] + a_j[b,p,:] + b1).
__global__ __launch_bounds__(256)
void k_relnet(const float* __restrict__ aI, const float* __restrict__ aJ,
              const float* __restrict__ b1,
              const _Float16* __restrict__ W2T, const float* __restrict__ b2,
              const _Float16* __restrict__ W3T, const float* __restrict__ b3,
              float* __restrict__ out) {
  extern __shared__ _Float16 smem[];
  _Float16* sH1 = smem;               // [MT][CD] swizzled
  _Float16* sH2 = smem + MT * CD;     // [MT][CD] swizzled   (total exactly 160KB)
  const int bImg = blockIdx.x >> 5, p = blockIdx.x & 31;
  const int tid = threadIdx.x;

  // Phase A: h1 tile into LDS (f16, swizzled), one 16B chunk per iteration
  {
    const float* ajrow = aJ + (size_t)(bImg * MT + p) * HD;
    for (int i = tid; i < MT * RCH; i += 256) {
      int q = i / RCH, c = i % RCH;
      const float* air = aI + (size_t)(bImg * MT + q) * HD + c * 8;
      float4 ai0 = *reinterpret_cast<const float4*>(air);
      float4 ai1 = *reinterpret_cast<const float4*>(air + 4);
      float4 aj0 = *reinterpret_cast<const float4*>(ajrow + c * 8);
      float4 aj1 = *reinterpret_cast<const float4*>(ajrow + c * 8 + 4);
      float4 bb0 = *reinterpret_cast<const float4*>(b1 + c * 8);
      float4 bb1 = *reinterpret_cast<const float4*>(b1 + c * 8 + 4);
      float x[8] = { ai0.x + aj0.x + bb0.x, ai0.y + aj0.y + bb0.y,
                     ai0.z + aj0.z + bb0.z, ai0.w + aj0.w + bb0.w,
                     ai1.x + aj1.x + bb1.x, ai1.y + aj1.y + bb1.y,
                     ai1.z + aj1.z + bb1.z, ai1.w + aj1.w + bb1.w };
      v8h h;
#pragma unroll
      for (int e = 0; e < 8; ++e) h[e] = (_Float16)(x[e] > 0.f ? x[e] : 0.f);
      *reinterpret_cast<v8h*>(sH1 + q * CD + ((c ^ (q & 15)) << 3)) = h;
    }
  }
  __syncthreads();

  const int wave = tid >> 5, lane = tid & 31;
  const int l16 = lane & 15, hi = lane >> 4;

  // Phase B: h2 = relu(h1 @ W2 + b2) -> LDS (swizzled)
  {
    const _Float16* a0base = sH1 + l16 * CD;
    const _Float16* a1base = a0base + 16 * CD;
    for (int g = 0; g < 5; ++g) {
      const int nb0 = wave * 160 + g * 32, nb1 = nb0 + 16;
      v8f acc00 = {}, acc01 = {}, acc10 = {}, acc11 = {};
      const _Float16* b0p = W2T + (size_t)(nb0 + l16) * HD + (hi << 4);
      const _Float16* b1p = b0p + (size_t)16 * HD;
#pragma unroll 2
      for (int kb = 0; kb < HD; kb += 32) {
        const int c0 = (kb >> 3) + (hi << 1);
        v16h a0 = load_a_swz(a0base, l16, c0);
        v16h a1 = load_a_swz(a1base, l16, c0);
        v16h b0 = load_b_frag(b0p + kb);
        v16h b1 = load_b_frag(b1p + kb);
        acc00 = WMMA_F16(a0, b0, acc00);
        acc01 = WMMA_F16(a0, b1, acc01);
        acc10 = WMMA_F16(a1, b0, acc10);
        acc11 = WMMA_F16(a1, b1, acc11);
      }
      const float bias0 = b2[nb0 + l16], bias1 = b2[nb1 + l16];
      const int cn0 = (nb0 + l16) >> 3, of0 = (nb0 + l16) & 7;
      const int cn1 = (nb1 + l16) >> 3, of1 = (nb1 + l16) & 7;
#pragma unroll
      for (int r = 0; r < 8; ++r) {
        const int s = hi * 8 + r;                 // row&15 for both m-tiles
        _Float16* p0 = sH2 + s * CD;              // rows 0..15
        _Float16* p1 = sH2 + (16 + s) * CD;       // rows 16..31
        float v;
        v = acc00[r] + bias0; p0[((cn0 ^ s) << 3) + of0] = (_Float16)(v > 0.f ? v : 0.f);
        v = acc01[r] + bias1; p0[((cn1 ^ s) << 3) + of1] = (_Float16)(v > 0.f ? v : 0.f);
        v = acc10[r] + bias0; p1[((cn0 ^ s) << 3) + of0] = (_Float16)(v > 0.f ? v : 0.f);
        v = acc11[r] + bias1; p1[((cn1 ^ s) << 3) + of1] = (_Float16)(v > 0.f ? v : 0.f);
      }
    }
  }
  __syncthreads();

  // Phase C: h3 = relu(h2 @ W3 + b3); column-sum over 32 rows; atomic into out[b]
  {
    const _Float16* a0base = sH2 + l16 * CD;
    const _Float16* a1base = a0base + 16 * CD;
    for (int g = 0; g < 5; ++g) {
      const int nb0 = wave * 160 + g * 32, nb1 = nb0 + 16;
      v8f acc00 = {}, acc01 = {}, acc10 = {}, acc11 = {};
      const _Float16* b0p = W3T + (size_t)(nb0 + l16) * HD + (hi << 4);
      const _Float16* b1p = b0p + (size_t)16 * HD;
#pragma unroll 2
      for (int kb = 0; kb < HD; kb += 32) {
        const int c0 = (kb >> 3) + (hi << 1);
        v16h a0 = load_a_swz(a0base, l16, c0);
        v16h a1 = load_a_swz(a1base, l16, c0);
        v16h b0 = load_b_frag(b0p + kb);
        v16h b1 = load_b_frag(b1p + kb);
        acc00 = WMMA_F16(a0, b0, acc00);
        acc01 = WMMA_F16(a0, b1, acc01);
        acc10 = WMMA_F16(a1, b0, acc10);
        acc11 = WMMA_F16(a1, b1, acc11);
      }
      const float bias0 = b3[nb0 + l16], bias1 = b3[nb1 + l16];
      float s0 = 0.f, s1 = 0.f;
#pragma unroll
      for (int r = 0; r < 8; ++r) {
        float v;
        v = acc00[r] + bias0; s0 += (v > 0.f ? v : 0.f);
        v = acc10[r] + bias0; s0 += (v > 0.f ? v : 0.f);
        v = acc01[r] + bias1; s1 += (v > 0.f ? v : 0.f);
        v = acc11[r] + bias1; s1 += (v > 0.f ? v : 0.f);
      }
      // lanes l and l+16 cover the two row-halves of each m-tile
      atomicAdd(out + (size_t)bImg * HD + nb0 + l16, s0);
      atomicAdd(out + (size_t)bImg * HD + nb1 + l16, s1);
    }
  }
}

extern "C" void kernel_launch(void* const* d_in, const int* in_sizes, int n_in,
                              void* d_out, int out_size, void* d_ws, size_t ws_size,
                              hipStream_t stream) {
  const float* roi = (const float*)d_in[0];   // [32,32,1280]
  const float* W1  = (const float*)d_in[1];   // [2560,1280]
  const float* b1  = (const float*)d_in[2];   // [1280]
  const float* W2  = (const float*)d_in[3];   // [1280,1280]
  const float* b2  = (const float*)d_in[4];
  const float* W3  = (const float*)d_in[5];
  const float* b3  = (const float*)d_in[6];
  float* out = (float*)d_out;                 // [32,1280]

  char* ws = (char*)d_ws;
  _Float16* roih = (_Float16*)ws; ws += (size_t)BM * CD * 2;         // 2.62 MB
  _Float16* W1T  = (_Float16*)ws; ws += (size_t)HD * 2 * CD * 2;     // 6.55 MB
  _Float16* W2T  = (_Float16*)ws; ws += (size_t)HD * HD * 2;         // 3.28 MB
  _Float16* W3T  = (_Float16*)ws; ws += (size_t)HD * HD * 2;         // 3.28 MB
  float*    aI   = (float*)ws;    ws += (size_t)BM * HD * 4;         // 5.24 MB
  float*    aJ   = (float*)ws;    ws += (size_t)BM * HD * 4;         // 5.24 MB

  k_cvt_f16<<<(BM * CD / 4 + 255) / 256, 256, 0, stream>>>(roi, roih, BM * CD / 4);
  k_transpose_f16<<<(2 * CD * HD + 255) / 256, 256, 0, stream>>>(W1, W1T, 2 * CD, HD);
  k_transpose_f16<<<(HD * HD + 255) / 256, 256, 0, stream>>>(W2, W2T, HD, HD);
  k_transpose_f16<<<(HD * HD + 255) / 256, 256, 0, stream>>>(W3, W3T, HD, HD);

  k_gemm1<<<64, 256, MT * CD * sizeof(_Float16), stream>>>(roih, W1T, aI, aJ);

  k_zero<<<(32 * HD + 255) / 256, 256, 0, stream>>>(out, 32 * HD);
  k_relnet<<<1024, 256, 2 * MT * CD * sizeof(_Float16), stream>>>(
      aI, aJ, b1, W2T, b2, W3T, b3, out);
}
